// Conv_Flow_49374944035429
// MI455X (gfx1250) — matrix operand order repaired and checked by reading, not compiled, not run on hardware
//
#include <hip/hip_runtime.h>
#include <hip/hip_bf16.h>
#include <math.h>

// ---------------------------------------------------------------------------
// CDNA5 (gfx1250) WMMA implementation of the Conv_Flow coupling layer.
//   * all GEMM work on v_wmma_f32_16x16x32_bf16 (bf16 A/B, f32 accum)
//   * weights pre-swizzled into contiguous per-(tap,kchunk) 4KB tiles so the
//     W stage is a flat global->LDS copy (async-to-LDS, ASYNCcnt tracked)
//   * X tiles staged transposed ([col][k]) so B fragments are 2x ds_load_b128
//   * proj (192->2784) fused with the RQ spline (182MB tensor never hits HBM)
// ---------------------------------------------------------------------------

typedef __attribute__((ext_vector_type(16))) __bf16 v16bf;
typedef __attribute__((ext_vector_type(2)))  __bf16 bf16x2;
typedef __attribute__((ext_vector_type(8)))  float  v8f;
typedef __attribute__((ext_vector_type(4)))  int    v4i;

#define TT 2048
#define BB 8
#define CCH 192
#define HALF 96
#define TAIL_F 5.0f
#define SQRT_C_F 13.856406460551018f

#define KSTR 40   // transposed X tile: per-column K stride (32 + 8), 80B = 16B multiple
#define PSTR 132  // f32 proj tile row stride (128 + 4)

// ---------------------------------------------------------------------------
// gfx1250 async global->LDS staging. Probe round 2 revealed the builtin's
// parameter type: pointer to 4xi32 in AS(1) (global) / AS(3) (LDS).
// ---------------------------------------------------------------------------
#if __has_builtin(__builtin_amdgcn_global_load_async_to_lds_b128)
#define HAVE_ASYNC_LDS 1
typedef __attribute__((address_space(1))) v4i gv4i_t;
typedef __attribute__((address_space(3))) v4i lv4i_t;
#endif

__device__ __forceinline__ void async_copy16(void* l, const void* g) {
#ifdef HAVE_ASYNC_LDS
  __builtin_amdgcn_global_load_async_to_lds_b128((gv4i_t*)g, (lv4i_t*)l, 0, 0);
#else
  *(uint4*)l = *(const uint4*)g;
#endif
}
__device__ __forceinline__ void async_join() {
#ifdef HAVE_ASYNC_LDS
#if __has_builtin(__builtin_amdgcn_s_wait_asynccnt)
  __builtin_amdgcn_s_wait_asynccnt(0);
#else
  asm volatile("s_wait_asynccnt 0" ::: "memory");
#endif
#endif
}

// ---------------------------------------------------------------------------
// Vectorized WMMA fragment loads (2x ds_load_b128 each).
// A 16x32 (MxK), contiguous 32-elem rows: lanes 0-15 M=lane elems
// {K0..7,K16..23}; lanes 16-31 elems {K8..15,K24..31}.
// B 32x16 (KxN) from transposed [col][k] tile: lanes 0-15 N=lane
// K {0..7,16..23}, lanes 16-31 K {8..15,24..31} (same order as A).
// ---------------------------------------------------------------------------
struct Q2 { uint4 a, b; };

__device__ __forceinline__ v16bf load_a_frag(const __bf16* lw, int lane) {
  const __bf16* row = lw + (lane & 15) * 32;
  int kb = (lane >> 4) << 3;  // 0 or 8
  Q2 q;
  q.a = *(const uint4*)(row + kb);
  q.b = *(const uint4*)(row + 16 + kb);
  return __builtin_bit_cast(v16bf, q);
}

__device__ __forceinline__ v16bf load_b_frag(const __bf16* lxcol, int lane) {
  const __bf16* p = lxcol + (lane & 15) * KSTR + ((lane >> 4) << 3);
  Q2 q;
  q.a = *(const uint4*)(p);
  q.b = *(const uint4*)(p + 16);
  return __builtin_bit_cast(v16bf, q);
}

__device__ __forceinline__ float softplusf(float v) {
  return (v > 20.f) ? v : log1pf(__expf(v));
}
__device__ __forceinline__ float mishf(float v) {
  return v * tanhf(softplusf(v));
}

// ---------------------------------------------------------------------------
// WMMA conv1d-as-GEMM. Wsw is pre-swizzled: [mt][tap][kc][ml(64)][kl(32)].
// Block: 256 threads (8 waves), tile M=64 x N=128.
// grid.x = B*(T/128), grid.y = M/64.
// ---------------------------------------------------------------------------
__global__ __launch_bounds__(256)
void wmma_conv_gemm(const float* __restrict__ X, const __bf16* __restrict__ Wsw,
                    const float* __restrict__ bias, const float* __restrict__ extra,
                    const float* __restrict__ mask,
                    const float* __restrict__ bng, const float* __restrict__ bnb,
                    const float* __restrict__ bnm, const float* __restrict__ bnv,
                    const float* __restrict__ resid, float* __restrict__ Y,
                    int KCH, int XC, int M, int TAPS, int dil) {
  __shared__ __align__(16) __bf16 lw[64 * 32];
  __shared__ __align__(16) __bf16 lx[128 * KSTR];

  const int tid  = threadIdx.x;
  const int lane = tid & 31;
  const int wv   = tid >> 5;
  const int wm   = wv & 3;       // M sub-tile (4 x 16)
  const int wn   = wv >> 2;      // N sub-tile (2 x 64)
  const int b     = blockIdx.x >> 4;
  const int tbase = (blockIdx.x & 15) * 128;
  const int mblk  = blockIdx.y * 64;

  // 32-bit offset bases (SADDR + voffset addressing)
  const float* __restrict__ Xb = X + (size_t)b * XC * TT;
  const float* __restrict__ mb = mask ? mask + (size_t)b * TT : nullptr;

  v8f acc[4] = {};
  const int nk = KCH >> 5;

  for (int tap = 0; tap < TAPS; ++tap) {
    const int tshift = (tap - (TAPS >> 1)) * dil;
    for (int kc = 0; kc < nk; ++kc) {
      // --- W tile: flat 4KB async copy (swizzled layout) ---
      const __bf16* wsrc =
          Wsw + ((((size_t)blockIdx.y * TAPS + tap) * nk + kc) << 11);
      async_copy16(&lw[tid * 8], &wsrc[tid * 8]);

      // --- X tile: 32 k x 128 t, f32 -> bf16x2, stored transposed [col][k] ---
#pragma unroll
      for (int j = 0; j < 8; ++j) {
        int idx = tid + j * 256;
        int kp = idx >> 7;          // k pair 0..15
        int cl = idx & 127;         // column
        int t = tbase + tshift + cl;
        int k0 = kc * 32 + kp * 2;
        float v0 = 0.f, v1 = 0.f;
        if (t >= 0 && t < TT) {
          v0 = Xb[k0 * TT + t];
          v1 = Xb[(k0 + 1) * TT + t];
          if (mb) { float mv = mb[t]; v0 *= mv; v1 *= mv; }
        }
        bf16x2 p = {(__bf16)v0, (__bf16)v1};
        *(bf16x2*)&lx[cl * KSTR + kp * 2] = p;
      }
      if (kc + 1 < nk)  // global_prefetch_b8
        __builtin_prefetch(&Xb[((kc + 1) * 32) * TT + tbase], 0, 1);
      async_join();
      __syncthreads();

      v16bf a = load_a_frag(&lw[wm * 16 * 32], lane);
#pragma unroll
      for (int s = 0; s < 4; ++s) {
        v16bf bf = load_b_frag(&lx[(wn * 64 + s * 16) * KSTR], lane);
        acc[s] = __builtin_amdgcn_wmma_f32_16x16x32_bf16(
            false, a, false, bf, (short)0, acc[s], false, false);
      }
      __syncthreads();
    }
  }

  // --- epilogue: bias / conditions / mish+BN / residual ---
  const float* __restrict__ eb = extra ? extra + (size_t)b * M * TT : nullptr;
  const float* __restrict__ rb = resid ? resid + (size_t)b * M * TT : nullptr;
  float* __restrict__ Yb = Y + (size_t)b * M * TT;
#pragma unroll
  for (int s = 0; s < 4; ++s) {
#pragma unroll
    for (int r = 0; r < 8; ++r) {
      int m = mblk + wm * 16 + ((lane >> 4) << 3) + r;
      int t = tbase + wn * 64 + s * 16 + (lane & 15);
      int oi = m * TT + t;
      float v = acc[s][r];
      if (bias)  v += bias[m];
      if (eb)    v += eb[oi];
      if (bng) {
        v = mishf(v);
        v = (v - bnm[m]) * rsqrtf(bnv[m] + 1e-5f) * bng[m] + bnb[m];
      }
      if (rb) v += rb[oi];
      Yb[oi] = v;
    }
  }
}

// ---------------------------------------------------------------------------
// Spline helpers
// ---------------------------------------------------------------------------
__device__ __forceinline__ void rq_knots(const float* u, float mc,
                                         float* cw, float* wd) {
  float mx = u[0];
#pragma unroll
  for (int j = 1; j < 10; ++j) mx = fmaxf(mx, u[j]);
  float e[10]; float s = 0.f;
#pragma unroll
  for (int j = 0; j < 10; ++j) { e[j] = __expf(u[j] - mx); s += e[j]; }
  float inv = 1.f / s;
  float cum = 0.f;
  cw[0] = -TAIL_F;
#pragma unroll
  for (int j = 0; j < 10; ++j) {
    float p = mc + (1.f - mc * 10.f) * e[j] * inv;
    cum += p;
    cw[j + 1] = 2.f * TAIL_F * cum - TAIL_F;
  }
  cw[10] = TAIL_F;
#pragma unroll
  for (int j = 0; j < 10; ++j) wd[j] = cw[j + 1] - cw[j];
}

// ---------------------------------------------------------------------------
// Fused proj GEMM (rows padded 29->32, swizzled [c][kc][32][32]) + RQ spline.
// grid.x = B*HALF, grid.y = T/128.
// ---------------------------------------------------------------------------
__global__ __launch_bounds__(256)
void proj_spline(const float* __restrict__ hid, const __bf16* __restrict__ Wp,
                 const float* __restrict__ pb, const float* __restrict__ mask,
                 const float* __restrict__ x, float* __restrict__ out,
                 float* __restrict__ logdet) {
  __shared__ __align__(16) __bf16 lw[32 * 32];
  __shared__ __align__(16) __bf16 lx[128 * KSTR];
  __shared__ float lp[32 * PSTR];

  const int tid = threadIdx.x, lane = tid & 31, wv = tid >> 5;
  const int wm = wv & 1, wn = wv >> 1;
  const int b = blockIdx.x / HALF, c = blockIdx.x % HALF;
  const int tbase = blockIdx.y * 128;

  const float* __restrict__ hb = hid + (size_t)b * CCH * TT;
  const float* __restrict__ mb = mask + (size_t)b * TT;

  v8f acc[2] = {};
  for (int kc = 0; kc < 6; ++kc) {
    // W tile: flat 2KB async copy
    const __bf16* wsrc = Wp + (((size_t)c * 6 + kc) << 10);
    if (tid < 128) async_copy16(&lw[tid * 8], &wsrc[tid * 8]);
    // X tile transposed
#pragma unroll
    for (int j = 0; j < 8; ++j) {
      int idx = tid + j * 256;
      int kp = idx >> 7, cl = idx & 127;
      int t = tbase + cl;
      int k0 = kc * 32 + kp * 2;
      float mv = mb[t];
      bf16x2 p = {(__bf16)(hb[k0 * TT + t] * mv),
                  (__bf16)(hb[(k0 + 1) * TT + t] * mv)};
      *(bf16x2*)&lx[cl * KSTR + kp * 2] = p;
    }
    async_join();
    __syncthreads();
    v16bf a = load_a_frag(&lw[wm * 16 * 32], lane);
#pragma unroll
    for (int s = 0; s < 2; ++s) {
      v16bf bf = load_b_frag(&lx[(wn * 32 + s * 16) * KSTR], lane);
      acc[s] = __builtin_amdgcn_wmma_f32_16x16x32_bf16(
          false, a, false, bf, (short)0, acc[s], false, false);
    }
    __syncthreads();
  }

  // deposit proj values (+bias) into LDS, rows = spline params, cols = t
#pragma unroll
  for (int s = 0; s < 2; ++s) {
#pragma unroll
    for (int r = 0; r < 8; ++r) {
      int row = wm * 16 + ((lane >> 4) << 3) + r;
      int col = wn * 32 + s * 16 + (lane & 15);
      float v = acc[s][r] + ((row < 29) ? pb[c * 29 + row] : 0.f);
      lp[row * PSTR + col] = v;
    }
  }
  __syncthreads();

  if (tid < 128) {
    int t = tbase + tid;
    float mk = mb[t];
    float uw[10], uh[10], ud[9];
#pragma unroll
    for (int j = 0; j < 10; ++j) uw[j] = lp[j * PSTR + tid] * mk / SQRT_C_F;
#pragma unroll
    for (int j = 0; j < 10; ++j) uh[j] = lp[(10 + j) * PSTR + tid] * mk / SQRT_C_F;
#pragma unroll
    for (int j = 0; j < 9; ++j) ud[j] = lp[(20 + j) * PSTR + tid] * mk;

    float x1 = x[((size_t)b * CCH + HALF + c) * TT + t];

    float cw[11], wd[10], ch[11], hh[10], dd[11];
    rq_knots(uw, 0.001f, cw, wd);
    rq_knots(uh, 0.001f, ch, hh);
    float cpad = __logf(__expf(0.999f) - 1.f);
    dd[0] = 0.001f + softplusf(cpad);
    dd[10] = dd[0];
#pragma unroll
    for (int j = 0; j < 9; ++j) dd[j + 1] = 0.001f + softplusf(ud[j]);

    float xc = fminf(fmaxf(x1, -TAIL_F), TAIL_F);
    int idx = 0;
#pragma unroll
    for (int j = 1; j < 10; ++j) if (xc >= cw[j]) idx = j;

    // register-resident gather (avoid scratch from dynamic indexing)
    float icw = cw[0], iw = wd[0], ich = ch[0], ih = hh[0], d0 = dd[0], d1 = dd[1];
#pragma unroll
    for (int j = 1; j < 10; ++j) {
      bool sel = (j == idx);
      icw = sel ? cw[j] : icw;  iw = sel ? wd[j] : iw;
      ich = sel ? ch[j] : ich;  ih = sel ? hh[j] : ih;
      d0  = sel ? dd[j] : d0;   d1 = sel ? dd[j + 1] : d1;
    }

    float idl = ih / iw;
    float th = (xc - icw) / iw;
    float om = 1.f - th;
    float t1m = th * om;
    float den = idl + (d0 + d1 - 2.f * idl) * t1m;
    float num = ih * (idl * th * th + d0 * t1m);
    float outv = ich + num / den;
    float dnum = idl * idl * (d1 * th * th + 2.f * idl * t1m + d0 * om * om);
    float lad = __logf(dnum) - 2.f * __logf(den);

    bool inside = (x1 >= -TAIL_F) && (x1 <= TAIL_F);
    outv = inside ? outv : x1;
    lad = inside ? lad : 0.f;

    out[((size_t)b * CCH + (HALF - 1 - c)) * TT + t] = outv * mk;
    atomicAdd(&logdet[b], lad * mk);
  }
}

// ---------------------------------------------------------------------------
// Weight pre-swizzle kernels: f32 -> bf16, tile-contiguous layouts
// ---------------------------------------------------------------------------
// dst[mt][tap][kc][ml(64)][kl(32)] <- src[(m,KCH,TAPS)]
__global__ void pack_conv_w(const float* __restrict__ src, __bf16* __restrict__ dst,
                            int KCH, int TAPS, int n) {
  int i = blockIdx.x * 256 + threadIdx.x;
  if (i >= n) return;
  int nk = KCH >> 5;
  int e  = i & 2047;
  int kl = e & 31, ml = e >> 5;
  int r  = i >> 11;
  int kc = r % nk; r /= nk;
  int tap = r % TAPS; int mt = r / TAPS;
  int m = mt * 64 + ml;
  int k = kc * 32 + kl;
  dst[i] = (__bf16)src[((size_t)m * KCH + k) * TAPS + tap];
}

// dst[c][kc][rl(32)][kl(32)], rows 29..31 zero  <- proj_w (2784,192)
__global__ void build_proj_w(const float* __restrict__ pw, __bf16* __restrict__ o) {
  int i = blockIdx.x * 256 + threadIdx.x;
  if (i >= HALF * 6 * 1024) return;
  int kl = i & 31;
  int rl = (i >> 5) & 31;
  int kc = (i >> 10) % 6;
  int c  = i / 6144;
  int k = kc * 32 + kl;
  float v = (rl < 29) ? pw[((size_t)(c * 29 + rl)) * CCH + k] : 0.f;
  o[i] = (__bf16)v;
}

__global__ void zero_logdet(float* ld) {
  if (threadIdx.x < BB) ld[threadIdx.x] = 0.f;
}

// x0 passthrough: out channel 191-c = x[b,c,t]*mask  (c in 0..95)
__global__ void copy_x0(const float* __restrict__ x, const float* __restrict__ mask,
                        float* __restrict__ out) {
  int i = blockIdx.x * 256 + threadIdx.x;
  if (i >= BB * HALF * TT) return;
  int t = i % TT; int r = i / TT; int c = r % HALF; int b = r / HALF;
  float v = x[((size_t)b * CCH + c) * TT + t] * mask[(size_t)b * TT + t];
  out[((size_t)b * CCH + (CCH - 1 - c)) * TT + t] = v;
}

// ---------------------------------------------------------------------------
extern "C" void kernel_launch(void* const* d_in, const int* in_sizes, int n_in,
                              void* d_out, int out_size, void* d_ws, size_t ws_size,
                              hipStream_t stream) {
  const float* x        = (const float*)d_in[0];
  const float* cond     = (const float*)d_in[1];
  const float* masks    = (const float*)d_in[2];
  const float* prenet_w = (const float*)d_in[3];
  const float* prenet_b = (const float*)d_in[4];
  const float* dds_w1   = (const float*)d_in[5];
  const float* dds_b1   = (const float*)d_in[6];
  const float* bn1g     = (const float*)d_in[7];
  const float* bn1b     = (const float*)d_in[8];
  const float* bn1m     = (const float*)d_in[9];
  const float* bn1v     = (const float*)d_in[10];
  const float* dds_w2   = (const float*)d_in[11];
  const float* dds_b2   = (const float*)d_in[12];
  const float* bn2g     = (const float*)d_in[13];
  const float* bn2b     = (const float*)d_in[14];
  const float* bn2m     = (const float*)d_in[15];
  const float* bn2v     = (const float*)d_in[16];
  const float* proj_w   = (const float*)d_in[17];
  const float* proj_b   = (const float*)d_in[18];

  float* out    = (float*)d_out;
  float* logdet = out + (size_t)BB * CCH * TT;

  // workspace carve-up
  float*  hid  = (float*)d_ws;                              // B*192*T f32
  float*  h1   = hid + (size_t)BB * CCH * TT;               // B*192*T f32
  __bf16* wpre = (__bf16*)(h1 + (size_t)BB * CCH * TT);     // 192*96 swizzled
  __bf16* w1b  = wpre + CCH * HALF;                         // 3*(192*192*3) swizzled
  __bf16* w2b  = w1b + 3 * CCH * CCH * 3;                   // 3*(192*192) swizzled
  __bf16* wpj  = w2b + 3 * CCH * CCH;                       // 96*6*1024 swizzled

  const int n_pre = CCH * HALF;           // 18432
  const int n_w1s = CCH * CCH * 3;        // 110592 per stage
  const int n_w2s = CCH * CCH;            // 36864 per stage
  const int n_pj  = HALF * 6 * 1024;      // 589824

  pack_conv_w<<<(n_pre + 255) / 256, 256, 0, stream>>>(prenet_w, wpre, HALF, 1, n_pre);
  for (int s = 0; s < 3; ++s) {
    pack_conv_w<<<(n_w1s + 255) / 256, 256, 0, stream>>>(
        dds_w1 + (size_t)s * n_w1s, w1b + (size_t)s * n_w1s, CCH, 3, n_w1s);
    pack_conv_w<<<(n_w2s + 255) / 256, 256, 0, stream>>>(
        dds_w2 + (size_t)s * n_w2s, w2b + (size_t)s * n_w2s, CCH, 1, n_w2s);
  }
  build_proj_w<<<(n_pj + 255) / 256, 256, 0, stream>>>(proj_w, wpj);
  zero_logdet<<<1, 32, 0, stream>>>(logdet);

  dim3 grid(BB * (TT / 128), CCH / 64);  // (128, 3)

  // prenet: hid = W_pre @ x0 + b + conditions
  wmma_conv_gemm<<<grid, 256, 0, stream>>>(
      x, wpre, prenet_b, cond, nullptr,
      nullptr, nullptr, nullptr, nullptr, nullptr, hid,
      HALF, CCH, CCH, 1, 1);

  int dil = 1;
  for (int s = 0; s < 3; ++s) {
    // h1 = BN(mish(conv3_dil(hid*mask)))
    wmma_conv_gemm<<<grid, 256, 0, stream>>>(
        hid, w1b + (size_t)s * n_w1s, dds_b1 + s * CCH, nullptr, masks,
        bn1g + s * CCH, bn1b + s * CCH, bn1m + s * CCH, bn1v + s * CCH,
        nullptr, h1, CCH, CCH, CCH, 3, dil);
    // hid = hid + BN(mish(conv1(h1)))
    wmma_conv_gemm<<<grid, 256, 0, stream>>>(
        h1, w2b + (size_t)s * n_w2s, dds_b2 + s * CCH, nullptr, nullptr,
        bn2g + s * CCH, bn2b + s * CCH, bn2m + s * CCH, bn2v + s * CCH,
        hid, hid, CCH, CCH, CCH, 1, 1);
    dil *= 3;
  }

  // x0 passthrough into flipped output
  int n_cp = BB * HALF * TT;
  copy_x0<<<(n_cp + 255) / 256, 256, 0, stream>>>(x, masks, out);

  // fused proj GEMM + spline
  dim3 grid3(BB * HALF, TT / 128);  // (768, 16)
  proj_spline<<<grid3, 256, 0, stream>>>(hid, wpj, proj_b, masks, x, out, logdet);
}
